// BinaryDiff_14534169330075
// MI455X (gfx1250) — matrix operand order repaired and checked by reading, not compiled, not run on hardware
//
#include <hip/hip_runtime.h>
#include <stdint.h>

// ---------------------------------------------------------------------------
// BinaryDiff fused GEMM for gfx1250 (MI455X, wave32, WMMA)
// out = x @ (base + coeff*(2*mask-1))   as ONE bf16 GEMM, f32 accumulate.
// ---------------------------------------------------------------------------

#define AS1 __attribute__((address_space(1)))
#define AS3 __attribute__((address_space(3)))

typedef __attribute__((ext_vector_type(16))) __bf16          v16bf;
typedef __attribute__((ext_vector_type(8)))  __bf16          v8bf;
typedef __attribute__((ext_vector_type(8)))  float           v8f;
typedef __attribute__((ext_vector_type(4)))  float           f32x4;
typedef __attribute__((ext_vector_type(8)))  unsigned short  u16x8;
typedef __attribute__((ext_vector_type(16))) unsigned short  u16x16;
typedef int v4i __attribute__((__vector_size__(16)));  // matches builtin param type

#if __has_builtin(__builtin_amdgcn_global_load_async_to_lds_b128)
#define HAS_ASYNC_LDS 1
#endif

// 16-byte global -> LDS copy. CDNA5 async path (ASYNCcnt tracked), with
// register-staging fallback if the toolchain lacks the builtin.
__device__ __forceinline__ void cp16_g2l(__bf16* l, const __bf16* g) {
#ifdef HAS_ASYNC_LDS
  AS3 v4i* lp = (AS3 v4i*)(uint32_t)(uintptr_t)l;  // low 32 bits of generic LDS ptr = LDS addr
  AS1 v4i* gp = (AS1 v4i*)(uintptr_t)g;
  __builtin_amdgcn_global_load_async_to_lds_b128(gp, lp, 0, 0);
#else
  *(uint4*)l = *(const uint4*)g;
#endif
}

__device__ __forceinline__ void wait_async() {
#ifdef HAS_ASYNC_LDS
#if __has_builtin(__builtin_amdgcn_s_wait_asynccnt)
  __builtin_amdgcn_s_wait_asynccnt(0);
#else
  asm volatile("s_wait_asynccnt 0" ::: "memory");
#endif
#endif
}

// Problem dims (from reference): B=4,S=2048 -> M=8192; DIN=K=4096; DOUT=N=4096.
constexpr int Mdim = 8192;
constexpr int Ndim = 4096;
constexpr int Kdim = 4096;

constexpr int BM  = 128;
constexpr int BN  = 256;
constexpr int BK  = 32;
constexpr int LDT = 40;                 // padded LDS row stride (elems): 80B, 16B-aligned rows
constexpr int KT  = Kdim / BK;          // 128 K-steps

// ---------------------------------------------------------------------------
// Kernel 1: W_t[o][i] = bf16( base[i][o] + coeff * (2*mask[i][o] - 1) )
// Tiled 32x32 transpose through LDS; coalesced; NT loads (read-once inputs),
// RT store (Wt must live in L2 for the GEMM).
// ---------------------------------------------------------------------------
__global__ void bd_build_wt(const float* __restrict__ base,
                            const int* __restrict__ mask,
                            const float* __restrict__ coeff,
                            unsigned short* __restrict__ wt_raw) {
  __bf16* wt = (__bf16*)wt_raw;
  __shared__ __bf16 tile[32][33];
  const float c = coeff[0];
  const int o0 = blockIdx.x * 32;
  const int i0 = blockIdx.y * 32;
  const int tx = threadIdx.x;   // along o (contiguous in base)
  const int ty = threadIdx.y;   // 0..7
#pragma unroll
  for (int r = 0; r < 4; ++r) {
    const int i = i0 + ty + r * 8;
    const size_t idx = (size_t)i * Ndim + o0 + tx;
    const float b = __builtin_nontemporal_load(base + idx);
    const int   m = __builtin_nontemporal_load(mask + idx);
    tile[ty + r * 8][tx] = (__bf16)(b + c * (float)(2 * m - 1));
  }
  __syncthreads();
#pragma unroll
  for (int r = 0; r < 4; ++r) {
    const int o = o0 + ty + r * 8;
    wt[(size_t)o * Kdim + i0 + tx] = tile[tx][ty + r * 8];  // contiguous along i
  }
}

// ---------------------------------------------------------------------------
// Kernel 2: x fp32 -> bf16, 8 elements / thread. NT loads for the fp32
// stream (read once); RT store so bf16 x stays resident in the 192MB L2.
// ---------------------------------------------------------------------------
__global__ void bd_cvt_x(const float* __restrict__ x,
                         unsigned short* __restrict__ xb_raw) {
  __bf16* xb = (__bf16*)xb_raw;
  const size_t i = ((size_t)blockIdx.x * blockDim.x + threadIdx.x) * 8;
  f32x4 a = __builtin_nontemporal_load((const f32x4*)(x + i));
  f32x4 b = __builtin_nontemporal_load((const f32x4*)(x + i) + 1);
  v8bf o;
  o[0] = (__bf16)a[0]; o[1] = (__bf16)a[1]; o[2] = (__bf16)a[2]; o[3] = (__bf16)a[3];
  o[4] = (__bf16)b[0]; o[5] = (__bf16)b[1]; o[6] = (__bf16)b[2]; o[7] = (__bf16)b[3];
  *(v8bf*)(xb + i) = o;
}

// ---------------------------------------------------------------------------
// Kernel 3: C[M,N] = A[M,K](bf16) * Bt[N,K](bf16), f32 accum via
// v_wmma_f32_16x16x32_bf16. 256 thr = 8 waves (2x4); wave tile 64x64 = 4x4
// WMMA tiles -> 16 wmma per 16 ds_load_b128 per K-step.
// Double-buffered LDS, async global->LDS staging.
// ---------------------------------------------------------------------------
__global__ __launch_bounds__(256)
void bd_gemm(const unsigned short* __restrict__ A_raw,
             const unsigned short* __restrict__ Bt_raw,
             float* __restrict__ C) {
  const __bf16* A  = (const __bf16*)A_raw;
  const __bf16* Bt = (const __bf16*)Bt_raw;

  __shared__ __bf16 sA[2][BM * LDT];   // 2 x 128 x 40 x 2B = 20 KiB
  __shared__ __bf16 sB[2][BN * LDT];   // 2 x 256 x 40 x 2B = 40 KiB  (60 KiB total)

  const int tid = threadIdx.x;
  const int m0  = blockIdx.y * BM;
  const int n0  = blockIdx.x * BN;

  // copy assignment: 4 x 16B segments per 32-elem (64B) row
  const int seg  = tid & 3;
  const int crow = tid >> 2;                         // 0..63
  const __bf16* gA = A  + (size_t)(m0 + crow) * Kdim + seg * 8;
  const __bf16* gB = Bt + (size_t)(n0 + crow) * Kdim + seg * 8;
  const int loff = crow * LDT + seg * 8;

  // wave/lane mapping for WMMA fragments
  const int lane = tid & 31;
  const int wv   = tid >> 5;          // 0..7
  const int wm   = wv >> 2;           // 0..1 : 64-row slab of BM=128
  const int wn   = wv & 3;            // 0..3 : 64-col slab of BN=256
  const int mr   = lane & 15;         // M (A) / N (B) index within 16
  const int kh   = lane >> 4;         // K-half selector

  v8f acc[4][4] = {};

  // stage copier: A rows crow, crow+64 ; B rows crow .. crow+192
#define STAGE_COPY(S, KTOFF)                                                  \
  do {                                                                        \
    const __bf16* ga = gA + (size_t)(KTOFF)*BK;                               \
    const __bf16* gb = gB + (size_t)(KTOFF)*BK;                               \
    cp16_g2l(&sA[S][loff],             ga);                                   \
    cp16_g2l(&sA[S][loff + 64 * LDT],  ga + (size_t)64 * Kdim);               \
    cp16_g2l(&sB[S][loff],             gb);                                   \
    cp16_g2l(&sB[S][loff + 64 * LDT],  gb + (size_t)64 * Kdim);               \
    cp16_g2l(&sB[S][loff + 128 * LDT], gb + (size_t)128 * Kdim);              \
    cp16_g2l(&sB[S][loff + 192 * LDT], gb + (size_t)192 * Kdim);              \
  } while (0)

  STAGE_COPY(0, 0);
  wait_async();
  __syncthreads();

  int buf = 0;
  for (int kt = 0; kt < KT; ++kt) {
    if (kt + 1 < KT) STAGE_COPY(buf ^ 1, kt + 1);

    // A fragments (16-bit A layout): lanes 0-15: K 0-7 & 16-23; lanes 16-31: +8
    v16bf af[4];
#pragma unroll
    for (int t = 0; t < 4; ++t) {
      const __bf16* p = &sA[buf][(wm * 64 + t * 16 + mr) * LDT + kh * 8];
      u16x8 lo = *(const u16x8*)p;
      u16x8 hi = *(const u16x8*)(p + 16);
      u16x16 w;
#pragma unroll
      for (int i = 0; i < 8; ++i) { w[i] = lo[i]; w[8 + i] = hi[i]; }
      af[t] = __builtin_bit_cast(v16bf, w);
    }
    // B fragments: lanes 0-15: K 0-7 & 16-23; lanes 16-31: +8 (N = mr)
    v16bf bfv[4];
#pragma unroll
    for (int u = 0; u < 4; ++u) {
      const __bf16* q = &sB[buf][(wn * 64 + u * 16 + mr) * LDT + kh * 8];
      u16x8 lo = *(const u16x8*)q;
      u16x8 hi = *(const u16x8*)(q + 16);
      u16x16 w;
#pragma unroll
      for (int i = 0; i < 8; ++i) { w[i] = lo[i]; w[8 + i] = hi[i]; }
      bfv[u] = __builtin_bit_cast(v16bf, w);
    }

#pragma unroll
    for (int t = 0; t < 4; ++t)
#pragma unroll
      for (int u = 0; u < 4; ++u)
        acc[t][u] = __builtin_amdgcn_wmma_f32_16x16x32_bf16(
            false, af[t], false, bfv[u], (short)0, acc[t][u], false, false);

    wait_async();
    __syncthreads();
    buf ^= 1;
  }
#undef STAGE_COPY

  // Epilogue: 32-bit C/D layout -> VGPR v: lanes0-15 M=v, lanes16-31 M=v+8.
  // Non-temporal stores: output is write-once, keep it out of L2.
#pragma unroll
  for (int t = 0; t < 4; ++t) {
#pragma unroll
    for (int u = 0; u < 4; ++u) {
      const int col   = n0 + wn * 64 + u * 16 + mr;
      const int rbase = m0 + wm * 64 + t * 16 + kh * 8;
      float* cp = C + (size_t)rbase * Ndim + col;
#pragma unroll
      for (int v = 0; v < 8; ++v)
        __builtin_nontemporal_store(acc[t][u][v], cp + (size_t)v * Ndim);
    }
  }
}

// ---------------------------------------------------------------------------
// Host launcher
// ---------------------------------------------------------------------------
extern "C" void kernel_launch(void* const* d_in, const int* in_sizes, int n_in,
                              void* d_out, int out_size, void* d_ws, size_t ws_size,
                              hipStream_t stream) {
  const float* x     = (const float*)d_in[0];
  const float* base  = (const float*)d_in[1];
  const int*   mask  = (const int*)d_in[2];
  const float* coeff = (const float*)d_in[3];
  float* out = (float*)d_out;

  // workspace layout: Wt bf16 [N,K] = 32 MiB, then x bf16 [M,K] = 64 MiB
  unsigned short* wt = (unsigned short*)d_ws;
  unsigned short* xb = (unsigned short*)((char*)d_ws + (size_t)Ndim * Kdim * 2);

  bd_build_wt<<<dim3(Ndim / 32, Kdim / 32), dim3(32, 8), 0, stream>>>(base, mask, coeff, wt);
  bd_cvt_x<<<dim3((unsigned)((size_t)Mdim * Kdim / 8 / 256)), dim3(256), 0, stream>>>(x, xb);
  bd_gemm<<<dim3(Ndim / BN, Mdim / BM), dim3(256), 0, stream>>>(xb, wt, out);

  (void)in_sizes; (void)n_in; (void)out_size; (void)ws_size;
}
